// RNNBase_34093450395676
// MI455X (gfx1250) — 8-bit matrix operand order repaired and checked by reading, compiled, not run on hardware
//
#include <hip/hip_runtime.h>
#include <cstdint>
#include <cstddef>


typedef __attribute__((ext_vector_type(8))) int v8i;

#define T_STEPS 1024
#define HID     1024
#define KDIM    1024
#define G4      4096
#define NTILES  256
#define KBLK    16
#define QMAXF   127.0f
#define EPSQ    1e-8f

__device__ __forceinline__ float sigmoidf_(float x) {
    return 1.0f / (1.0f + __expf(-x));
}

__device__ __forceinline__ int pack4(const float* p, float inv_s) {
    int d = 0;
#pragma unroll
    for (int b = 0; b < 4; ++b) {
        float q = rintf(p[b] * inv_s);
        q = fminf(fmaxf(q, -127.0f), 127.0f);
        d |= (((int)q) & 0xFF) << (8 * b);
    }
    return d;
}

__global__ void wabsmax_kernel(const float* w0, const float* w1,
                               const float* w2, const float* w3,
                               unsigned* wmax_bits) {
    const int tensor = blockIdx.x >> 6;
    const int chunk  = blockIdx.x & 63;
    const float* w = (tensor == 0) ? w0 : (tensor == 1) ? w1
                   : (tensor == 2) ? w2 : w3;
    const int base = chunk * 65536;
    float m = 0.0f;
    for (int i = threadIdx.x; i < 65536; i += 256)
        m = fmaxf(m, fabsf(w[base + i]));
    __shared__ float red[256];
    red[threadIdx.x] = m;
    __syncthreads();
    for (int s = 128; s > 0; s >>= 1) {
        if (threadIdx.x < s)
            red[threadIdx.x] = fmaxf(red[threadIdx.x], red[threadIdx.x + s]);
        __syncthreads();
    }
    if (threadIdx.x == 0)
        atomicMax(&wmax_bits[tensor], __float_as_uint(red[0]));
}

__global__ void wpack_kernel(const float* w0, const float* w1,
                             const float* w2, const float* w3,
                             const unsigned* wmax_bits, uint8_t* wq) {
    const int item = blockIdx.x * 256 + threadIdx.x;
    const int tensor = item >> 17;
    const int rem    = item & 131071;
    const int n_tile = rem >> 9;
    const int kb     = (rem >> 5) & 15;
    const int lane   = rem & 31;
    const float* w = (tensor == 0) ? w0 : (tensor == 1) ? w1
                   : (tensor == 2) ? w2 : w3;
    const float s = fmaxf(__uint_as_float(wmax_bits[tensor]), EPSQ) / QMAXF;
    const float inv_s = 1.0f / s;
    const int n = n_tile * 16 + (lane & 15);
    v8i out = {0, 0, 0, 0, 0, 0, 0, 0};
#pragma unroll
    for (int v = 0; v < 8; ++v) {
        const int k0 = kb * 64 + (v >> 1) * 16 + ((lane >> 4) << 3) + (v & 1) * 4;
        out[v] = pack4(&w[(size_t)n * KDIM + k0], inv_s);
    }
    *(v8i*)(wq + (size_t)tensor * 4194304 +
            (size_t)(((n_tile * 16 + kb) * 32 + lane)) * 32) = out;
}

__global__ void actmax_kernel(const float* x, float* xscale) {
    const int t = blockIdx.x;
    const float* xt = x + (size_t)t * 16384;
    float m = 0.0f;
    for (int i = threadIdx.x; i < 16384; i += 256)
        m = fmaxf(m, fabsf(xt[i]));
    __shared__ float red[256];
    red[threadIdx.x] = m;
    __syncthreads();
    for (int s = 128; s > 0; s >>= 1) {
        if (threadIdx.x < s)
            red[threadIdx.x] = fmaxf(red[threadIdx.x], red[threadIdx.x + s]);
        __syncthreads();
    }
    if (threadIdx.x == 0)
        xscale[t] = fmaxf(red[0], EPSQ) / QMAXF;
}

__global__ void actpack_kernel(const float* x, const float* xscale, uint8_t* xq) {
    const int item = blockIdx.x * 256 + threadIdx.x;
    const int t    = item >> 9;
    const int kb   = (item >> 5) & 15;
    const int lane = item & 31;
    const float inv_s = 1.0f / xscale[t];
    const float* xt = x + (size_t)t * 16384 + (size_t)(lane & 15) * KDIM;
    v8i out = {0, 0, 0, 0, 0, 0, 0, 0};
#pragma unroll
    for (int v = 0; v < 8; ++v) {
        const int k0 = kb * 64 + (v >> 1) * 16 + ((lane >> 4) << 3) + (v & 1) * 4;
        out[v] = pack4(xt + k0, inv_s);
    }
    *(v8i*)(xq + (size_t)t * 16384 + (size_t)((kb * 32 + lane)) * 32) = out;
}

__global__ void pregate_kernel(const uint8_t* xq, const uint8_t* wq,
                               const unsigned* wmax_bits, const float* xscale,
                               const float* b_ih, const float* b_hh,
                               float* preg) {
    const int wave = threadIdx.x >> 5;
    const int lane = threadIdx.x & 31;
    const int lin  = blockIdx.x * 8 + wave;
    const int t      = lin >> 8;
    const int n_tile = lin & 255;
    const v8i* A = (const v8i*)(xq + (size_t)t * 16384);
    const v8i* B = (const v8i*)wq + (size_t)n_tile * (KBLK * 32);
    v8i acc = {0, 0, 0, 0, 0, 0, 0, 0};
    for (int kb = 0; kb < KBLK; ++kb) {
        const v8i av = A[kb * 32 + lane];
        const v8i bv = B[kb * 32 + lane];
        __builtin_prefetch(&B[(kb + 1) * 32 + lane], 0, 1);
        acc = __builtin_amdgcn_wmma_i32_16x16x64_iu8(true, av, true, bv, acc,
                                                     false, false);
    }
    const float wsc = fmaxf(__uint_as_float(wmax_bits[0]), EPSQ) / QMAXF;
    const float sc  = xscale[t] * wsc;
    const int col = n_tile * 16 + (lane & 15);
    const int mh  = (lane >> 4) * 8;
#pragma unroll
    for (int vg = 0; vg < 8; ++vg) {
        const int m = vg + mh;
        preg[((size_t)t * 16 + m) * G4 + col] =
            (float)acc[vg] * sc + b_ih[col] + b_hh[col];
    }
}

__global__ void init_hc_kernel(const float* h0, const float* c0,
                               float* hbuf, float* cbuf, float* hmax) {
    float m = 0.0f;
    for (int i = threadIdx.x; i < 16384; i += 256) {
        const float h = h0[i];
        hbuf[i] = h;
        cbuf[i] = c0[i];
        m = fmaxf(m, fabsf(h));
    }
    __shared__ float red[256];
    red[threadIdx.x] = m;
    __syncthreads();
    for (int s = 128; s > 0; s >>= 1) {
        if (threadIdx.x < s)
            red[threadIdx.x] = fmaxf(red[threadIdx.x], red[threadIdx.x + s]);
        __syncthreads();
    }
    if (threadIdx.x == 0) hmax[0] = red[0];
}

__global__ void lstm_seq_kernel(const float* __restrict__ preg,
                                const uint8_t* __restrict__ whq,
                                const unsigned* __restrict__ wmax_bits,
                                float* hmax, float* hbuf, float* cbuf,
                                float* yout, float* hn_out, float* cn_out,
                                unsigned* bar, int nsteps) {
    __shared__ v8i aq[512];
    __shared__ unsigned bar_target;
    const int tid  = threadIdx.x;
    const int lane = tid & 31;
    const int wave = tid >> 5;
    const int gw   = blockIdx.x * 8 + wave;
    if (tid == 0) bar_target = 0;
    __syncthreads();

    const float wsc = fmaxf(__uint_as_float(wmax_bits[0]), EPSQ) / QMAXF;
    const v8i* W = (const v8i*)whq;
    const int col = gw * 16 + (lane & 15);
    const int mh  = (lane >> 4) * 8;

    for (int t = 0; t < nsteps; ++t) {
        const float* hread = hbuf + (t & 1) * 16384;
        const float hs = fmaxf(hmax[t], EPSQ) / QMAXF;
        const float inv_hs = 1.0f / hs;

        for (int item = tid; item < 512; item += 256) {
            const int kb = item >> 5;
            const int l  = item & 31;
            const float* hp = hread + (size_t)(l & 15) * HID;
            v8i out = {0, 0, 0, 0, 0, 0, 0, 0};
#pragma unroll
            for (int v = 0; v < 8; ++v) {
                const int k0 = kb * 64 + (v >> 1) * 16 + ((l >> 4) << 3) + (v & 1) * 4;
                out[v] = pack4(hp + k0, inv_hs);
            }
            aq[item] = out;
        }
        __syncthreads();

        v8i acc0 = {0, 0, 0, 0, 0, 0, 0, 0};
        v8i acc1 = {0, 0, 0, 0, 0, 0, 0, 0};
        v8i acc2 = {0, 0, 0, 0, 0, 0, 0, 0};
        v8i acc3 = {0, 0, 0, 0, 0, 0, 0, 0};
        for (int kb = 0; kb < KBLK; ++kb) {
            const v8i av = aq[kb * 32 + lane];
            const v8i b0 = W[(size_t)(((0 * 64 + gw) * 16 + kb) * 32 + lane)];
            const v8i b1 = W[(size_t)(((1 * 64 + gw) * 16 + kb) * 32 + lane)];
            const v8i b2 = W[(size_t)(((2 * 64 + gw) * 16 + kb) * 32 + lane)];
            const v8i b3 = W[(size_t)(((3 * 64 + gw) * 16 + kb) * 32 + lane)];
            acc0 = __builtin_amdgcn_wmma_i32_16x16x64_iu8(true, av, true, b0, acc0, false, false);
            acc1 = __builtin_amdgcn_wmma_i32_16x16x64_iu8(true, av, true, b1, acc1, false, false);
            acc2 = __builtin_amdgcn_wmma_i32_16x16x64_iu8(true, av, true, b2, acc2, false, false);
            acc3 = __builtin_amdgcn_wmma_i32_16x16x64_iu8(true, av, true, b3, acc3, false, false);
        }

        const float gsc = hs * wsc;
        float* hwrite = hbuf + ((t + 1) & 1) * 16384;
        float lmax = 0.0f;
        const size_t prow_base = (size_t)t * 16 * G4;
#pragma unroll
        for (int vg = 0; vg < 8; ++vg) {
            const int m = vg + mh;
            const size_t pr = prow_base + (size_t)m * G4 + col;
            const float iv = sigmoidf_((float)acc0[vg] * gsc + preg[pr]);
            const float fv = sigmoidf_((float)acc1[vg] * gsc + preg[pr + 1024]);
            const float gv = tanhf((float)acc2[vg] * gsc + preg[pr + 2048]);
            const float ov = sigmoidf_((float)acc3[vg] * gsc + preg[pr + 3072]);
            const int ci = m * HID + col;
            const float c_new = fv * cbuf[ci] + iv * gv;
            cbuf[ci] = c_new;
            const float h_new = ov * tanhf(c_new);
            hwrite[ci] = h_new;
            yout[(size_t)t * 16384 + ci] = h_new;
            lmax = fmaxf(lmax, fabsf(h_new));
            if (t == nsteps - 1) { hn_out[ci] = h_new; cn_out[ci] = c_new; }
        }

        for (int off = 16; off > 0; off >>= 1)
            lmax = fmaxf(lmax, __shfl_xor(lmax, off, 32));
        if (lane == 0)
            atomicMax((unsigned*)&hmax[t + 1], __float_as_uint(lmax));

        __syncthreads();
        if (tid == 0) {
            __threadfence();
            const unsigned tgt = bar_target + gridDim.x;
            bar_target = tgt;
            atomicAdd(bar, 1u);
            while (__hip_atomic_load(bar, __ATOMIC_ACQUIRE,
                                     __HIP_MEMORY_SCOPE_AGENT) < tgt)
                __builtin_amdgcn_s_sleep(1);
        }
        __syncthreads();
    }
}

extern "C" void kernel_launch(void* const* d_in, const int* in_sizes, int n_in,
                              void* d_out, int out_size, void* d_ws,
                              size_t ws_size, hipStream_t stream) {
    (void)in_sizes; (void)n_in; (void)out_size; (void)ws_size;

    const float* x     = (const float*)d_in[0];
    const float* h0    = (const float*)d_in[1];
    const float* c0    = (const float*)d_in[2];
    const float* w_ih0 = (const float*)d_in[3];
    const float* w_hh0 = (const float*)d_in[4];
    const float* b_ih0 = (const float*)d_in[5];
    const float* b_hh0 = (const float*)d_in[6];
    const float* w_ih1 = (const float*)d_in[7];
    const float* w_hh1 = (const float*)d_in[8];
    const float* b_ih1 = (const float*)d_in[9];
    const float* b_hh1 = (const float*)d_in[10];

    uint8_t* ws = (uint8_t*)d_ws;
    const size_t OFF_WQ     = 0;
    const size_t OFF_WMAX   = (size_t)4 * 4194304;
    const size_t OFF_XSCALE = OFF_WMAX + 256;
    const size_t OFF_BAR    = OFF_XSCALE + 4096;
    const size_t OFF_HMAX   = OFF_BAR + 256;
    const size_t OFF_XQ     = OFF_HMAX + 8448;
    const size_t OFF_PREG   = OFF_XQ + (size_t)T_STEPS * 16384;
    const size_t OFF_Y0     = OFF_PREG + (size_t)T_STEPS * 16 * G4 * 4;
    const size_t OFF_HBUF   = OFF_Y0 + (size_t)T_STEPS * 16 * HID * 4;
    const size_t OFF_CBUF   = OFF_HBUF + (size_t)2 * 16384 * 4;

    unsigned* wmax  = (unsigned*)(ws + OFF_WMAX);
    float* xscale   = (float*)(ws + OFF_XSCALE);
    unsigned* bar   = (unsigned*)(ws + OFF_BAR);
    float* hmax     = (float*)(ws + OFF_HMAX);
    uint8_t* xq     = ws + OFF_XQ;
    float* preg     = (float*)(ws + OFF_PREG);
    float* y0       = (float*)(ws + OFF_Y0);
    float* hbuf     = (float*)(ws + OFF_HBUF);
    float* cbuf     = (float*)(ws + OFF_CBUF);

    hipMemsetAsync(ws + OFF_WMAX, 0, OFF_XQ - OFF_WMAX, stream);

    wabsmax_kernel<<<256, 256, 0, stream>>>(w_ih0, w_hh0, w_ih1, w_hh1, wmax);
    wpack_kernel<<<2048, 256, 0, stream>>>(w_ih0, w_hh0, w_ih1, w_hh1, wmax,
                                           ws + OFF_WQ);

    float* y_final = (float*)d_out;
    float* hn_base = (float*)d_out + (size_t)T_STEPS * 16 * HID;
    float* cn_base = hn_base + (size_t)2 * 16384;

    for (int l = 0; l < 2; ++l) {
        const float* xin = l ? y0 : x;
        const uint8_t* wq_ih = ws + (size_t)(l ? 2 : 0) * 4194304;
        const uint8_t* wq_hh = ws + (size_t)(l ? 3 : 1) * 4194304;
        const unsigned* wmax_ih = wmax + (l ? 2 : 0);
        const unsigned* wmax_hh = wmax + (l ? 3 : 1);
        const float* b_ih = l ? b_ih1 : b_ih0;
        const float* b_hh = l ? b_hh1 : b_hh0;
        float* hmax_l = hmax + (size_t)l * (T_STEPS + 1);
        unsigned* bar_l = bar + l;
        float* yout = l ? y_final : y0;

        actmax_kernel<<<T_STEPS, 256, 0, stream>>>(xin, xscale);
        actpack_kernel<<<2048, 256, 0, stream>>>(xin, xscale, xq);
        pregate_kernel<<<T_STEPS * 32, 256, 0, stream>>>(xq, wq_ih, wmax_ih,
                                                         xscale, b_ih, b_hh,
                                                         preg);
        init_hc_kernel<<<1, 256, 0, stream>>>(h0 + (size_t)l * 16384,
                                              c0 + (size_t)l * 16384,
                                              hbuf, cbuf, hmax_l);
        lstm_seq_kernel<<<8, 256, 0, stream>>>(preg, wq_hh, wmax_hh, hmax_l,
                                               hbuf, cbuf, yout,
                                               hn_base + (size_t)l * 16384,
                                               cn_base + (size_t)l * 16384,
                                               bar_l, T_STEPS);
    }
}
